// TrafficGNN_31971736551686
// MI455X (gfx1250) — matrix operand order repaired and checked by reading, not compiled, not run on hardware
//
#include <hip/hip_runtime.h>
#include <hip/hip_bf16.h>
#include <math.h>

typedef __attribute__((ext_vector_type(16))) __bf16 v16bf;
typedef __attribute__((ext_vector_type(8)))  __bf16 v8bf;
typedef __attribute__((ext_vector_type(8)))  float  v8f;

#define HEADS 4
#define HID 64
#define FEAT 256           // HEADS * HID
#define NEG_SLOPE 0.2f
#define GEMM_WAVES 8
#define AGG_WAVES 4

__device__ __forceinline__ float lrelu(float x) {
  // slope < 1 -> max(x, slope*x) == leaky_relu
  return fmaxf(x, NEG_SLOPE * x);
}
__device__ __forceinline__ float gelu_tanh(float x) {
  float x3 = x * x * x;
  return 0.5f * x * (1.0f + tanhf(0.7978845608028654f * (x + 0.044715f * x3)));
}

// ---------------- CSR build (once; graph shared by all 3 layers) ----------------
__global__ void k_deg_hist(const int* __restrict__ ei, int E, int Nn, int* __restrict__ deg) {
  int e = blockIdx.x * blockDim.x + threadIdx.x;
  if (e >= E + Nn) return;
  int d = (e < E) ? ei[E + e] : (e - E);   // ei[1][e] is dst; tail = self loops
  atomicAdd(&deg[d], 1);
}

__global__ void k_scan(const int* __restrict__ deg, int* __restrict__ rowptr, int Nn) {
  __shared__ int part[1024];
  int tid = threadIdx.x;
  int chunk = (Nn + 1023) >> 10;
  int b = tid * chunk;
  int e = min(b + chunk, Nn);
  int s = 0;
  for (int i = b; i < e; ++i) s += deg[i];
  part[tid] = s;
  __syncthreads();
  for (int off = 1; off < 1024; off <<= 1) {
    int v = (tid >= off) ? part[tid - off] : 0;
    __syncthreads();
    part[tid] += v;
    __syncthreads();
  }
  int run = (tid == 0) ? 0 : part[tid - 1];
  for (int i = b; i < e; ++i) { rowptr[i] = run; run += deg[i]; }
  if (tid == 1023) rowptr[Nn] = part[1023];
}

__global__ void k_csr_scatter(const int* __restrict__ ei, int E, int Nn,
                              int* __restrict__ cursor, int* __restrict__ csr_src) {
  int e = blockIdx.x * blockDim.x + threadIdx.x;
  if (e >= E + Nn) return;
  int s, d;
  if (e < E) { s = ei[e]; d = ei[E + e]; } else { s = e - E; d = s; }
  int pos = atomicAdd(&cursor[d], 1);
  csr_src[pos] = s;
}

// ---------------- precision conversion / weight packing ----------------
__global__ void k_f32_to_bf16(const float* __restrict__ in, __bf16* __restrict__ out, int n) {
  int i = blockIdx.x * blockDim.x + threadIdx.x;
  if (i < n) out[i] = (__bf16)in[i];
}

// W [K,256] f32 -> WT [256,K] bf16 so each WMMA B-lane reads 16 contiguous K values
__global__ void k_pack_wt(const float* __restrict__ W, __bf16* __restrict__ WT, int K) {
  int i = blockIdx.x * blockDim.x + threadIdx.x;
  if (i >= K * FEAT) return;
  int k = i >> 8;
  int n = i & 255;
  WT[n * K + k] = (__bf16)W[k * FEAT + n];
}

// ---------------- WMMA GEMM: C[M,256] = A[M,K](bf16) * W[K,256] ----------------
// one 16x16 tile per wave; K compile-time so the 16x16x32 k-loop fully unrolls.
template <int K>
__global__ void k_gemm_wmma(const __bf16* __restrict__ A, const __bf16* __restrict__ BT,
                            float* __restrict__ C) {
  int wave = threadIdx.x >> 5;
  int lane = threadIdx.x & 31;
  int row0 = blockIdx.x << 4;
  int col_tile = blockIdx.y * GEMM_WAVES + wave;   // 0..15
  int mn   = lane & 15;                            // row for A, col for B
  int half = lane >> 4;
  // A (16x32), contiguous k order used by this code: lanes 0-15 hold K 0-15, lanes 16-31 hold K 16-31
  const __bf16* arow = A + (size_t)(row0 + mn) * K + half * 16;
  // B (32x16), same contiguous k order as A: lanes 0-15 hold K 0-15 of column n, lanes 16-31 hold K 16-31
  const __bf16* brow = BT + (size_t)((col_tile << 4) + mn) * K + half * 16;
  v8f acc = {};
#pragma unroll
  for (int k = 0; k < K; k += 32) {
    v8bf a0 = *(const v8bf*)(arow + k);
    v8bf a1 = *(const v8bf*)(arow + k + 8);
    v16bf av;
#pragma unroll
    for (int t = 0; t < 8; ++t) { av[t] = a0[t]; av[t + 8] = a1[t]; }
    v16bf bv = *(const v16bf*)(brow + k);
    acc = __builtin_amdgcn_wmma_f32_16x16x32_bf16(false, av, false, bv, (short)0, acc,
                                                  false, false);
  }
  // C/D layout: VGPR v -> row v + 8*(lane>=16), col = lane&15
  float* out = C + (size_t)(row0 + (half << 3)) * FEAT + (col_tile << 4) + mn;
#pragma unroll
  for (int v = 0; v < 8; ++v) out[(size_t)v * FEAT] = acc[v];
}

// ---------------- per-node attention coefficients es/ed [N,4] ----------------
__global__ void k_attn_coef(const float* __restrict__ Hf, const float* __restrict__ a_src,
                            const float* __restrict__ a_dst, float* __restrict__ es,
                            float* __restrict__ ed, int Nn) {
  int t = blockIdx.x * blockDim.x + threadIdx.x;
  if (t >= Nn * HEADS) return;
  int node = t >> 2, h = t & 3;
  const float* hp = Hf + (size_t)node * FEAT + h * HID;
  const float* as = a_src + h * HID;
  const float* ad = a_dst + h * HID;
  float s = 0.f, d = 0.f;
#pragma unroll 8
  for (int c = 0; c < HID; ++c) { float v = hp[c]; s = fmaf(v, as[c], s); d = fmaf(v, ad[c], d); }
  es[t] = s;
  ed[t] = d;
}

// ---------------- GAT core: per-dst-node softmax + aggregation (one wave/node) ----
// Fuses head-mean + bias + GELU + f32->bf16 (writes next layer's GEMM input).
__global__ void k_gat_aggregate(const float* __restrict__ Hf, const float* __restrict__ es,
                                const float* __restrict__ ed, const int* __restrict__ rowptr,
                                const int* __restrict__ csr_src, const float* __restrict__ bias,
                                __bf16* __restrict__ Xn, int Nn) {
  int wave = threadIdx.x >> 5;
  int lane = threadIdx.x & 31;
  int node = blockIdx.x * AGG_WAVES + wave;
  if (node >= Nn) return;
  int beg = rowptr[node], end = rowptr[node + 1];
  float4 edv = *(const float4*)(ed + (size_t)node * 4);

  // pass 1: per-head running max over incoming edges (strided by lane)
  float m0 = -1e30f, m1 = -1e30f, m2 = -1e30f, m3 = -1e30f;
  for (int j = beg + lane; j < end; j += 32) {
    int s = csr_src[j];
    float4 e = *(const float4*)(es + (size_t)s * 4);
    m0 = fmaxf(m0, lrelu(e.x + edv.x));
    m1 = fmaxf(m1, lrelu(e.y + edv.y));
    m2 = fmaxf(m2, lrelu(e.z + edv.z));
    m3 = fmaxf(m3, lrelu(e.w + edv.w));
  }
#pragma unroll
  for (int off = 16; off > 0; off >>= 1) {
    m0 = fmaxf(m0, __shfl_xor(m0, off, 32));
    m1 = fmaxf(m1, __shfl_xor(m1, off, 32));
    m2 = fmaxf(m2, __shfl_xor(m2, off, 32));
    m3 = fmaxf(m3, __shfl_xor(m3, off, 32));
  }

  // pass 2: per-head sum(exp(e - m))
  float d0 = 0.f, d1 = 0.f, d2 = 0.f, d3 = 0.f;
  for (int j = beg + lane; j < end; j += 32) {
    int s = csr_src[j];
    float4 e = *(const float4*)(es + (size_t)s * 4);
    d0 += __expf(lrelu(e.x + edv.x) - m0);
    d1 += __expf(lrelu(e.y + edv.y) - m1);
    d2 += __expf(lrelu(e.z + edv.z) - m2);
    d3 += __expf(lrelu(e.w + edv.w) - m3);
  }
#pragma unroll
  for (int off = 16; off > 0; off >>= 1) {
    d0 += __shfl_xor(d0, off, 32);
    d1 += __shfl_xor(d1, off, 32);
    d2 += __shfl_xor(d2, off, 32);
    d3 += __shfl_xor(d3, off, 32);
  }

  // pass 3: lane l owns channels l*8..l*8+7 -> head = l/8; whole wave walks edges.
  int h = lane >> 3;
  float mh  = (h == 0) ? m0 : (h == 1) ? m1 : (h == 2) ? m2 : m3;
  float dh  = (h == 0) ? d0 : (h == 1) ? d1 : (h == 2) ? d2 : d3;
  float edh = (h == 0) ? edv.x : (h == 1) ? edv.y : (h == 2) ? edv.z : edv.w;
  float inv = 1.0f / (dh + 1e-16f);
  float acc[8] = {0.f, 0.f, 0.f, 0.f, 0.f, 0.f, 0.f, 0.f};
  for (int j = beg; j < end; ++j) {
    int s = csr_src[j];
    float a = __expf(lrelu(es[(size_t)s * 4 + h] + edh) - mh) * inv;
    const float4* hp = (const float4*)(Hf + (size_t)s * FEAT + lane * 8);
    float4 p0 = hp[0], p1 = hp[1];
    acc[0] = fmaf(a, p0.x, acc[0]); acc[1] = fmaf(a, p0.y, acc[1]);
    acc[2] = fmaf(a, p0.z, acc[2]); acc[3] = fmaf(a, p0.w, acc[3]);
    acc[4] = fmaf(a, p1.x, acc[4]); acc[5] = fmaf(a, p1.y, acc[5]);
    acc[6] = fmaf(a, p1.z, acc[6]); acc[7] = fmaf(a, p1.w, acc[7]);
  }
  // head-mean: channels c, c+64, c+128, c+192 live in lanes l, l^8, l^16, l^24
#pragma unroll
  for (int i = 0; i < 8; ++i) {
    acc[i] += __shfl_xor(acc[i], 8, 32);
    acc[i] += __shfl_xor(acc[i], 16, 32);
  }
  if (lane < 8) {
#pragma unroll
    for (int i = 0; i < 8; ++i) {
      int c = lane * 8 + i;
      float r = gelu_tanh(acc[i] * 0.25f + bias[c]);
      Xn[(size_t)node * HID + c] = (__bf16)r;
    }
  }
}

// ---------------- final fc (64 -> 1) + sigmoid ----------------
__global__ void k_fc_sigmoid(const __bf16* __restrict__ X, const float* __restrict__ Wf,
                             const float* __restrict__ bf, float* __restrict__ out, int Nn) {
  int n = blockIdx.x * blockDim.x + threadIdx.x;
  if (n >= Nn) return;
  float s = bf[0];
  const __bf16* xp = X + (size_t)n * HID;
#pragma unroll 8
  for (int c = 0; c < HID; ++c) s = fmaf((float)xp[c], Wf[c], s);
  out[n] = 1.0f / (1.0f + __expf(-s));
}

extern "C" void kernel_launch(void* const* d_in, const int* in_sizes, int n_in,
                              void* d_out, int out_size, void* d_ws, size_t ws_size,
                              hipStream_t stream) {
  const float* x = (const float*)d_in[0];
  const int* ei = (const int*)d_in[1];
  const float* Wl[3]   = {(const float*)d_in[2], (const float*)d_in[6], (const float*)d_in[10]};
  const float* Asrc[3] = {(const float*)d_in[3], (const float*)d_in[7], (const float*)d_in[11]};
  const float* Adst[3] = {(const float*)d_in[4], (const float*)d_in[8], (const float*)d_in[12]};
  const float* Bl[3]   = {(const float*)d_in[5], (const float*)d_in[9], (const float*)d_in[13]};
  const float* fcW = (const float*)d_in[14];
  const float* fcB = (const float*)d_in[15];
  float* out = (float*)d_out;

  int Nn = in_sizes[0] / 128;   // IN_C = 128
  int E  = in_sizes[1] / 2;     // edge_index is [2, E]
  int Etot = E + Nn;            // + self loops

  size_t off = 0;
  auto take = [&](size_t bytes) -> void* {
    off = (off + 255) & ~(size_t)255;
    void* p = (char*)d_ws + off;
    off += bytes;
    return p;
  };
  int*    deg     = (int*)take((size_t)Nn * 4);
  int*    rowptr  = (int*)take((size_t)(Nn + 1) * 4);
  int*    cursor  = (int*)take((size_t)Nn * 4);
  int*    csr_src = (int*)take((size_t)Etot * 4);
  __bf16* WT      = (__bf16*)take((size_t)FEAT * 128 * 2);
  __bf16* XB      = (__bf16*)take((size_t)Nn * 128 * 2);
  float*  Hf      = (float*)take((size_t)Nn * FEAT * 4);
  float*  es      = (float*)take((size_t)Nn * HEADS * 4);
  float*  ed      = (float*)take((size_t)Nn * HEADS * 4);
  if (off > ws_size) return;    // insufficient workspace

  const int tb = 256;

  // CSR by destination (built once; reused for all 3 layers)
  hipMemsetAsync(deg, 0, (size_t)Nn * 4, stream);
  k_deg_hist<<<(Etot + tb - 1) / tb, tb, 0, stream>>>(ei, E, Nn, deg);
  k_scan<<<1, 1024, 0, stream>>>(deg, rowptr, Nn);
  hipMemcpyAsync(cursor, rowptr, (size_t)Nn * 4, hipMemcpyDeviceToDevice, stream);
  k_csr_scatter<<<(Etot + tb - 1) / tb, tb, 0, stream>>>(ei, E, Nn, cursor, csr_src);

  // layer-1 input -> bf16
  k_f32_to_bf16<<<(Nn * 128 + tb - 1) / tb, tb, 0, stream>>>(x, XB, Nn * 128);

  for (int L = 0; L < 3; ++L) {
    int K = (L == 0) ? 128 : HID;
    k_pack_wt<<<(K * FEAT + tb - 1) / tb, tb, 0, stream>>>(Wl[L], WT, K);
    dim3 g(Nn / 16, (FEAT / 16) / GEMM_WAVES);   // N = 50000 is a multiple of 16
    if (K == 128)
      k_gemm_wmma<128><<<g, 32 * GEMM_WAVES, 0, stream>>>(XB, WT, Hf);
    else
      k_gemm_wmma<64><<<g, 32 * GEMM_WAVES, 0, stream>>>(XB, WT, Hf);
    k_attn_coef<<<(Nn * HEADS + tb - 1) / tb, tb, 0, stream>>>(Hf, Asrc[L], Adst[L], es, ed, Nn);
    k_gat_aggregate<<<(Nn + AGG_WAVES - 1) / AGG_WAVES, 32 * AGG_WAVES, 0, stream>>>(
        Hf, es, ed, rowptr, csr_src, Bl[L], XB, Nn);
  }

  k_fc_sigmoid<<<(Nn + tb - 1) / tb, tb, 0, stream>>>(XB, fcW, fcB, out, Nn);
}
